// LSTM_51110110822977
// MI455X (gfx1250) — matrix operand order repaired and checked by reading, not compiled, not run on hardware
//
#include <hip/hip_runtime.h>
#include <math.h>

// ---------------------------------------------------------------------------
// LSTM (2 layers, T=2048, B=32, I=256, H=512) + FC(1024) + log_softmax
// MI455X / gfx1250: wave32, WMMA 16x16x32 bf16 -> f32.
// Per-layer persistent kernel, 32 WGs each owning a 16-wide h-slice
// (64 gate columns), weights packed in LDS, A=[x_t|h_prev] staged in LDS
// via GLOBAL_LOAD_ASYNC_TO_LDS_B128 (ASYNCcnt), ping-pong h buffer through
// L2 + one device-wide barrier per timestep.
// ---------------------------------------------------------------------------

#define T_STEPS 2048
#define BATCH   32
#define IN0     256
#define HID     512
#define GATES   2048   // 4*HID
#define OUTD    1024

typedef __attribute__((ext_vector_type(16))) __bf16 bf16x16;
typedef __attribute__((ext_vector_type(8)))  float  f32x8;

union ABFrag { uint4 u[2]; bf16x16 v; };

#if defined(__has_builtin)
# if __has_builtin(__builtin_amdgcn_global_load_async_to_lds_b128)
#  define ASYNC_LDS 1
# endif
# if __has_builtin(__builtin_amdgcn_s_wait_asynccnt)
#  define WAIT_ASYNC_BUILTIN 1
# endif
#endif

#if defined(ASYNC_LDS)
typedef int v4i_t __attribute__((vector_size(16)));
typedef __attribute__((address_space(1))) v4i_t g_v4i;   // global (AS1)
typedef __attribute__((address_space(3))) v4i_t l_v4i;   // LDS (AS3)
static __device__ __forceinline__ void asyncCopy16(const void* g, void* l) {
    __builtin_amdgcn_global_load_async_to_lds_b128((g_v4i*)g, (l_v4i*)l, 0, 0);
}
#endif

static __device__ __forceinline__ void waitAsync0() {
#if defined(ASYNC_LDS)
# if defined(WAIT_ASYNC_BUILTIN)
    __builtin_amdgcn_s_wait_asynccnt(0);
# else
    asm volatile("s_wait_asynccnt 0x0" ::: "memory");
# endif
#endif
}

static __device__ __forceinline__ unsigned short f2bf(float f) {
    unsigned u = __float_as_uint(f);
    unsigned r = (u + 0x7FFFu + ((u >> 16) & 1u)) >> 16;   // RNE
    return (unsigned short)r;
}
static __device__ __forceinline__ float sigf(float x) {
    return 1.0f / (1.0f + expf(-x));
}

// ---------------- device-wide split barrier --------------------------------
static __device__ void gridBarrier(volatile unsigned* cnt, volatile unsigned* gen,
                                   unsigned nwg) {
    __syncthreads();
    if (threadIdx.x == 0) {
        __threadfence();
        unsigned g = *gen;
        unsigned a = atomicAdd((unsigned*)cnt, 1u) + 1u;
        if (a == nwg) {
            *cnt = 0u;
            __threadfence();
            atomicAdd((unsigned*)gen, 1u);
        } else {
            while (*gen == g) { __builtin_amdgcn_s_sleep(2); }
        }
        __threadfence();
    }
    __syncthreads();
}

// ---------------- small prep kernels ---------------------------------------
__global__ void k_cvt_bf16(const float* __restrict__ s, unsigned short* __restrict__ d,
                           long long n) {
    long long i = (long long)blockIdx.x * blockDim.x + threadIdx.x;
    long long st = (long long)gridDim.x * blockDim.x;
    for (; i < n; i += st) d[i] = f2bf(s[i]);
}

__global__ void k_bias_sum(const float* __restrict__ a, const float* __restrict__ b,
                           float* __restrict__ o, int n) {
    int i = blockIdx.x * blockDim.x + threadIdx.x;
    if (i < n) o[i] = a[i] + b[i];
}

__global__ void k_init_layer(const float* __restrict__ h0l,
                             unsigned short* __restrict__ hbuf0,
                             unsigned* __restrict__ cnt) {
    int i = blockIdx.x * blockDim.x + threadIdx.x;
    if (i < BATCH * HID) hbuf0[i] = f2bf(h0l[i]);
    if (i == 0) *cnt = 0u;
}

// ---------------- persistent recurrent layer kernel ------------------------
// grid = 32 WGs x 256 threads (8 waves). WG wg owns h-slice [wg*16, wg*16+16)
// => gate columns q*HID + wg*16 + hc for q in 0..3.
// LDS: Wlds[64][K] bf16, Xlds[32][K] bf16, gates[32][64] f32, cstate[32][16] f32.
template <int KX>
__global__ void k_lstm_layer(const unsigned short* __restrict__ xseq,
                             const unsigned short* __restrict__ wih,  // [GATES][KX] bf16
                             const unsigned short* __restrict__ whh,  // [GATES][HID] bf16
                             const float* __restrict__ bias,          // [GATES]
                             const float* __restrict__ c0l,           // [BATCH*HID]
                             unsigned short* __restrict__ hbufs,      // 2 x BATCH*HID bf16
                             unsigned short* __restrict__ hseq_out,   // [T][B][HID] or null
                             unsigned* bar_cnt, unsigned* bar_gen) {
    extern __shared__ char smem[];
    constexpr int K = KX + HID;
    char*  Ws     = smem;                                   // 64*K*2
    char*  Xs     = Ws + (size_t)64 * K * 2;                // 32*K*2
    float* gatesS = (float*)(Xs + (size_t)32 * K * 2);      // 32*64
    float* cst    = gatesS + 32 * 64;                       // 32*16

    const int wg  = blockIdx.x;
    const int tid = threadIdx.x;
    const int w    = tid >> 5;
    const int lane = tid & 31;
    const int laneM = lane & 15;
    const bool hi   = lane >= 16;
    const int m = w & 1;     // M-tile (batch rows m*16..)
    const int q = w >> 1;    // gate 0..3

    // ---- pack owned weight columns into LDS (once) ----
    constexpr int u4Ih = (KX * 2) / 16, u4Hh = (HID * 2) / 16;
    constexpr int u4Col = u4Ih + u4Hh;
    for (int i = tid; i < 64 * u4Col; i += 256) {
        int c = i / u4Col, r = i % u4Col;
        int gq = c >> 4, hc = c & 15;
        int gcol = gq * HID + wg * 16 + hc;
        const uint4* src = (r < u4Ih)
            ? ((const uint4*)wih) + (size_t)gcol * u4Ih + r
            : ((const uint4*)whh) + (size_t)gcol * u4Hh + (r - u4Ih);
#if defined(ASYNC_LDS)
        asyncCopy16(src, ((uint4*)Ws) + (size_t)c * u4Col + r);
#else
        ((uint4*)Ws)[(size_t)c * u4Col + r] = *src;
#endif
    }
    // ---- load c0 slice ----
    for (int i = tid; i < 32 * 16; i += 256) {
        int b = i >> 4, hc = i & 15;
        cst[i] = c0l[b * HID + wg * 16 + hc];
    }
    waitAsync0();

    constexpr int u4XRow = (KX * 2) / 16, u4HRow = (HID * 2) / 16;
    constexpr int u4Row = u4XRow + u4HRow;

    const char* Arow = Xs + (size_t)(m * 16 + laneM) * K * 2;
    const char* Brow = Ws + (size_t)(q * 16 + laneM) * K * 2;
    const int aShift = hi ? 32 : 0;   // K-half offset for A (16 elems)
    const int bShift = hi ? 32 : 0;   // K-half offset for B (16 elems)

    for (int t = 0; t < T_STEPS; ++t) {
        const unsigned short* hprev = hbufs + (size_t)(t & 1) * BATCH * HID;
        unsigned short*       hnext = hbufs + (size_t)((t + 1) & 1) * BATCH * HID;
        const unsigned short* xt = xseq + (size_t)t * BATCH * KX;

        // ---- stage A = [x_t | h_prev] into LDS (async to LDS, ASYNCcnt) ----
        for (int i = tid; i < 32 * u4Row; i += 256) {
            int b = i / u4Row, r = i % u4Row;
            const uint4* src = (r < u4XRow)
                ? ((const uint4*)xt) + (size_t)b * u4XRow + r
                : ((const uint4*)hprev) + (size_t)b * u4HRow + (r - u4XRow);
#if defined(ASYNC_LDS)
            asyncCopy16(src, ((uint4*)Xs) + i);
#else
            ((uint4*)Xs)[i] = *src;
#endif
        }
        if (t + 1 < T_STEPS)  // gfx1250 global_prefetch_b8 for next timestep's x
            __builtin_prefetch(xseq + (size_t)(t + 1) * BATCH * KX + tid * 8, 0, 1);
        waitAsync0();
        __syncthreads();

        // ---- GEMM: C[16x16] += A[16x32] * B[32x16] over K ----
        f32x8 acc = {0.f, 0.f, 0.f, 0.f, 0.f, 0.f, 0.f, 0.f};
        #pragma unroll 4
        for (int kb = 0; kb < K; kb += 32) {
            ABFrag a, b;
            a.u[0] = *(const uint4*)(Arow + kb * 2 + aShift);
            a.u[1] = *(const uint4*)(Arow + kb * 2 + aShift + 16);
            b.u[0] = *(const uint4*)(Brow + kb * 2 + bShift);
            b.u[1] = *(const uint4*)(Brow + kb * 2 + bShift + 16);
            acc = __builtin_amdgcn_wmma_f32_16x16x32_bf16(
                false, a.v, false, b.v, (short)0, acc, false, false);
        }
        // scatter C tile into gates LDS: row m*16+M, col q*16+laneM
        #pragma unroll
        for (int j = 0; j < 8; ++j) {
            int M = j + (hi ? 8 : 0);
            gatesS[(m * 16 + M) * 64 + q * 16 + laneM] = acc[j];
        }
        __syncthreads();

        // ---- elementwise gate math on the owned h-slice ----
        for (int i = tid; i < 512; i += 256) {
            int b = i >> 4, hc = i & 15;
            int g = wg * 16 + hc;
            float iv = gatesS[b * 64 + 0  + hc] + bias[0 * HID + g];
            float fv = gatesS[b * 64 + 16 + hc] + bias[1 * HID + g];
            float gv = gatesS[b * 64 + 32 + hc] + bias[2 * HID + g];
            float ov = gatesS[b * 64 + 48 + hc] + bias[3 * HID + g];
            float c = sigf(fv) * cst[i] + sigf(iv) * tanhf(gv);
            float h = sigf(ov) * tanhf(c);
            cst[i] = c;
            unsigned short hb = f2bf(h);
            hnext[b * HID + g] = hb;
            if (hseq_out) hseq_out[((size_t)t * BATCH + b) * HID + g] = hb;
        }
        gridBarrier(bar_cnt, bar_gen, gridDim.x);
    }
}

// ---------------- FC GEMM: logits[32x1024] = h2[32x512] * fc_w^T + fc_b ----
__global__ void k_fc(const unsigned short* __restrict__ h2,   // [32][512] bf16
                     const unsigned short* __restrict__ fcw,  // [1024][512] bf16
                     const float* __restrict__ fcb,
                     float* __restrict__ logits) {
    const int tid = threadIdx.x;
    const int w = tid >> 5, lane = tid & 31;
    const int laneM = lane & 15;
    const bool hi = lane >= 16;
    const int m = w & 1;
    const int nt = blockIdx.x * 4 + (w >> 1);   // 16 blocks * 4 = 64 N tiles

    const char* Arow = (const char*)h2 + (size_t)(m * 16 + laneM) * HID * 2;
    const char* Brow = (const char*)fcw + (size_t)(nt * 16 + laneM) * HID * 2;
    const int aShift = hi ? 32 : 0;
    const int bShift = hi ? 32 : 0;

    f32x8 acc = {0.f, 0.f, 0.f, 0.f, 0.f, 0.f, 0.f, 0.f};
    #pragma unroll 4
    for (int kb = 0; kb < HID; kb += 32) {
        ABFrag a, b;
        a.u[0] = *(const uint4*)(Arow + kb * 2 + aShift);
        a.u[1] = *(const uint4*)(Arow + kb * 2 + aShift + 16);
        b.u[0] = *(const uint4*)(Brow + kb * 2 + bShift);
        b.u[1] = *(const uint4*)(Brow + kb * 2 + bShift + 16);
        acc = __builtin_amdgcn_wmma_f32_16x16x32_bf16(
            false, a.v, false, b.v, (short)0, acc, false, false);
    }
    float bv = fcb[nt * 16 + laneM];
    #pragma unroll
    for (int j = 0; j < 8; ++j) {
        int M = j + (hi ? 8 : 0);
        logits[(size_t)(m * 16 + M) * OUTD + nt * 16 + laneM] = acc[j] + bv;
    }
}

// ---------------- log_softmax: one block per row ---------------------------
__global__ void k_logsoftmax(const float* __restrict__ logits, float* __restrict__ out) {
    __shared__ float red[256];
    const int row = blockIdx.x, tid = threadIdx.x;
    const float* r = logits + (size_t)row * OUTD;

    float mx = -1e30f;
    for (int c = tid; c < OUTD; c += 256) mx = fmaxf(mx, r[c]);
    red[tid] = mx; __syncthreads();
    for (int s = 128; s > 0; s >>= 1) {
        if (tid < s) red[tid] = fmaxf(red[tid], red[tid + s]);
        __syncthreads();
    }
    mx = red[0]; __syncthreads();

    float sum = 0.f;
    for (int c = tid; c < OUTD; c += 256) sum += expf(r[c] - mx);
    red[tid] = sum; __syncthreads();
    for (int s = 128; s > 0; s >>= 1) {
        if (tid < s) red[tid] += red[tid + s];
        __syncthreads();
    }
    float ls = logf(red[0]);
    for (int c = tid; c < OUTD; c += 256)
        out[(size_t)row * OUTD + c] = r[c] - mx - ls;
}

// ---------------- host side ------------------------------------------------
extern "C" void kernel_launch(void* const* d_in, const int* in_sizes, int n_in,
                              void* d_out, int out_size, void* d_ws, size_t ws_size,
                              hipStream_t stream) {
    const float* x     = (const float*)d_in[0];
    const float* h0    = (const float*)d_in[1];
    const float* c0    = (const float*)d_in[2];
    const float* w_ih0 = (const float*)d_in[3];
    const float* w_hh0 = (const float*)d_in[4];
    const float* b_ih0 = (const float*)d_in[5];
    const float* b_hh0 = (const float*)d_in[6];
    const float* w_ih1 = (const float*)d_in[7];
    const float* w_hh1 = (const float*)d_in[8];
    const float* b_ih1 = (const float*)d_in[9];
    const float* b_hh1 = (const float*)d_in[10];
    const float* fc_w  = (const float*)d_in[11];
    const float* fc_b  = (const float*)d_in[12];

    char* ws = (char*)d_ws;
    size_t off = 0;
    auto alloc = [&](size_t bytes) { char* p = ws + off; off = (off + bytes + 255) & ~(size_t)255; return p; };

    unsigned short* xbf    = (unsigned short*)alloc((size_t)T_STEPS * BATCH * IN0 * 2);
    unsigned short* h1seq  = (unsigned short*)alloc((size_t)T_STEPS * BATCH * HID * 2);
    unsigned short* wih0b  = (unsigned short*)alloc((size_t)GATES * IN0 * 2);
    unsigned short* whh0b  = (unsigned short*)alloc((size_t)GATES * HID * 2);
    unsigned short* wih1b  = (unsigned short*)alloc((size_t)GATES * HID * 2);
    unsigned short* whh1b  = (unsigned short*)alloc((size_t)GATES * HID * 2);
    unsigned short* fcwb   = (unsigned short*)alloc((size_t)OUTD * HID * 2);
    float*          b0     = (float*)alloc((size_t)GATES * 4);
    float*          b1     = (float*)alloc((size_t)GATES * 4);
    unsigned short* hbufs  = (unsigned short*)alloc((size_t)2 * BATCH * HID * 2);
    float*          logits = (float*)alloc((size_t)BATCH * OUTD * 4);
    unsigned*       bar    = (unsigned*)alloc(256);   // [0]=cnt, [1]=gen

    // -- conversions --
    k_cvt_bf16<<<2048, 256, 0, stream>>>(x, xbf, (long long)T_STEPS * BATCH * IN0);
    k_cvt_bf16<<<512,  256, 0, stream>>>(w_ih0, wih0b, (long long)GATES * IN0);
    k_cvt_bf16<<<512,  256, 0, stream>>>(w_hh0, whh0b, (long long)GATES * HID);
    k_cvt_bf16<<<512,  256, 0, stream>>>(w_ih1, wih1b, (long long)GATES * HID);
    k_cvt_bf16<<<512,  256, 0, stream>>>(w_hh1, whh1b, (long long)GATES * HID);
    k_cvt_bf16<<<512,  256, 0, stream>>>(fc_w,  fcwb,  (long long)OUTD * HID);
    k_bias_sum<<<(GATES + 255) / 256, 256, 0, stream>>>(b_ih0, b_hh0, b0, GATES);
    k_bias_sum<<<(GATES + 255) / 256, 256, 0, stream>>>(b_ih1, b_hh1, b1, GATES);

    const int K0 = IN0 + HID, K1 = HID + HID;
    const size_t lds0 = (size_t)64 * K0 * 2 + (size_t)32 * K0 * 2 + 32 * 64 * 4 + 32 * 16 * 4;
    const size_t lds1 = (size_t)64 * K1 * 2 + (size_t)32 * K1 * 2 + 32 * 64 * 4 + 32 * 16 * 4;

    // -- layer 0 --
    k_init_layer<<<64, 256, 0, stream>>>(h0, hbufs, bar);
    k_lstm_layer<IN0><<<32, 256, lds0, stream>>>(xbf, wih0b, whh0b, b0, c0,
                                                 hbufs, h1seq, bar, bar + 1);
    // -- layer 1 --
    k_init_layer<<<64, 256, 0, stream>>>(h0 + BATCH * HID, hbufs, bar);
    k_lstm_layer<HID><<<32, 256, lds1, stream>>>(h1seq, wih1b, whh1b, b1,
                                                 c0 + BATCH * HID, hbufs, nullptr,
                                                 bar, bar + 1);
    // T even => final h2 lives in hbufs[0]
    k_fc<<<16, 256, 0, stream>>>(hbufs, fcwb, fc_b, logits);
    k_logsoftmax<<<BATCH, 256, 0, stream>>>(logits, (float*)d_out);

    (void)in_sizes; (void)n_in; (void)out_size; (void)ws_size;
}
